// MultibandFrameAttention_63934883168451
// MI455X (gfx1250) — matrix operand order repaired and checked by reading, not compiled, not run on hardware
//
#include <hip/hip_runtime.h>

// MI455X / gfx1250, wave32. bf16 split-precision WMMA pipeline.
// Workspace layout (bytes), total ~336 MB:
//   [0,100663296)        Yq/Yk/Yv fp32 (33.5MB each)  -- reused later for Aout
//   [100663296,201326592) Qh,Ql,Kh,Kl,Vh,Vl bf16 planes (16.8MB each)
//   [201326592,201588736) relT hi/lo bf16 (512x128)
//   [201588736,335806464) Ph,Pl bf16 planes (67MB each)

typedef __attribute__((ext_vector_type(16))) __bf16 v16bf;
typedef __attribute__((ext_vector_type(8)))  float  v8f;

__device__ __forceinline__ unsigned short f32_to_bf16(float f) {
  unsigned x = __float_as_uint(f);
  unsigned r = x + 0x7FFFu + ((x >> 16) & 1u);   // round-to-nearest-even
  return (unsigned short)(r >> 16);
}

__device__ __forceinline__ void split_bf16(float f, unsigned short &hi, unsigned short &lo) {
  unsigned short h = f32_to_bf16(f);
  float fh = __uint_as_float(((unsigned)h) << 16);
  hi = h;
  lo = f32_to_bf16(f - fh);
}

__device__ __forceinline__ v8f wmma_bf16_pairs(const unsigned *a, const unsigned *b, v8f c) {
  union U { unsigned u[8]; v16bf v; };
  U A, B;
#pragma unroll
  for (int i = 0; i < 8; ++i) { A.u[i] = a[i]; B.u[i] = b[i]; }
  // v_wmma_f32_16x16x32_bf16: (neg_a, A, neg_b, B, c_mod, C, reuse_a, reuse_b)
  return __builtin_amdgcn_wmma_f32_16x16x32_bf16(false, A.v, false, B.v, (short)0, c, false, false);
}

// 16-bit A 16x32 fragment: lane m = lane&15;
// lanes 0-15: K=0..15 (2/VGPR), lanes 16-31: K=16..31.
__device__ __forceinline__ void load_frag_A(const unsigned short (*P)[34], int row0, int lane,
                                            unsigned *dst) {
  int m  = row0 + (lane & 15);
  int kh = (lane >> 4) << 4;  // 0 or 16
#pragma unroll
  for (int v = 0; v < 8; ++v) {
    int k = kh + (v << 1);
    dst[v] = *(const unsigned *)&P[m][k];
  }
}

// 16-bit B 32x16 fragment: lane n = lane&15; lanes 0-15 K=0..15 (2/VGPR), lanes 16-31 K=16..31.
__device__ __forceinline__ void load_frag_B(const unsigned short (*P)[34], int row0, int lane,
                                            unsigned *dst) {
  int n  = row0 + (lane & 15);
  int kh = (lane >> 4) << 4;  // 0 or 16
#pragma unroll
  for (int v = 0; v < 8; ++v) {
    int k = kh + (v << 1);
    dst[v] = *(const unsigned *)&P[n][k];
  }
}

// One BK=32 step of the 128x128 block: each wave does 64x32 (4 m-tiles x 2 n-tiles),
// 3 WMMAs per tile for the bf16 hi/lo split => 24 v_wmma per call.
__device__ __forceinline__ void mma_block(const unsigned short (*Ah)[34], const unsigned short (*Al)[34],
                                          const unsigned short (*Bh)[34], const unsigned short (*Bl)[34],
                                          int wm, int wn, int lane, v8f acc[4][2]) {
  unsigned bh[2][8], bl[2][8];
#pragma unroll
  for (int nt = 0; nt < 2; ++nt) {
    load_frag_B(Bh, 32 * wn + 16 * nt, lane, bh[nt]);
    load_frag_B(Bl, 32 * wn + 16 * nt, lane, bl[nt]);
  }
#pragma unroll
  for (int mt = 0; mt < 4; ++mt) {
    unsigned ah[8], al[8];
    load_frag_A(Ah, 64 * wm + 16 * mt, lane, ah);
    load_frag_A(Al, 64 * wm + 16 * mt, lane, al);
#pragma unroll
    for (int nt = 0; nt < 2; ++nt) {
      acc[mt][nt] = wmma_bf16_pairs(al, bh[nt], acc[mt][nt]);  // lo*hi
      acc[mt][nt] = wmma_bf16_pairs(ah, bl[nt], acc[mt][nt]);  // hi*lo
      acc[mt][nt] = wmma_bf16_pairs(ah, bh[nt], acc[mt][nt]);  // hi*hi
    }
  }
}

__device__ __forceinline__ void zero_acc(v8f acc[4][2]) {
#pragma unroll
  for (int i = 0; i < 4; ++i)
#pragma unroll
    for (int j = 0; j < 2; ++j) {
      v8f z;
#pragma unroll
      for (int e = 0; e < 8; ++e) z[e] = 0.0f;
      acc[i][j] = z;
    }
}

// ---------------------------------------------------------------------------
// Kernel 1/6: C[m,n] = sum_k A[m,k] * W[n,k]  (NT GEMM, fp32 in, fp32 out)
// grid: (N/128, M/128, nmat); W/C selected by blockIdx.z
// ---------------------------------------------------------------------------
__global__ __launch_bounds__(256) void gemm_xwT_kernel(
    const float *__restrict__ A,
    const float *__restrict__ W0, const float *__restrict__ W1, const float *__restrict__ W2,
    float *__restrict__ C0, float *__restrict__ C1, float *__restrict__ C2,
    int N, int K) {
  __shared__ __align__(8) unsigned short Ah[128][34], Al[128][34], Bh[128][34], Bl[128][34];
  const float *W = (blockIdx.z == 0) ? W0 : (blockIdx.z == 1) ? W1 : W2;
  float *C = (blockIdx.z == 0) ? C0 : (blockIdx.z == 1) ? C1 : C2;
  const int tid = threadIdx.x, lane = tid & 31, wid = tid >> 5;
  const int wm = wid & 1, wn = wid >> 1;
  const size_t m0 = (size_t)blockIdx.y * 128, n0 = (size_t)blockIdx.x * 128;
  v8f acc[4][2];
  zero_acc(acc);

  for (int kt = 0; kt < K; kt += 32) {
    __syncthreads();
#pragma unroll 1
    for (int p = tid; p < 128 * 8; p += 256) {
      int r = p >> 3, c = (p & 7) << 2;
      float4 fa = *(const float4 *)(A + (m0 + r) * (size_t)K + kt + c);
      float4 fw = *(const float4 *)(W + (n0 + r) * (size_t)K + kt + c);
      split_bf16(fa.x, Ah[r][c + 0], Al[r][c + 0]);
      split_bf16(fa.y, Ah[r][c + 1], Al[r][c + 1]);
      split_bf16(fa.z, Ah[r][c + 2], Al[r][c + 2]);
      split_bf16(fa.w, Ah[r][c + 3], Al[r][c + 3]);
      split_bf16(fw.x, Bh[r][c + 0], Bl[r][c + 0]);
      split_bf16(fw.y, Bh[r][c + 1], Bl[r][c + 1]);
      split_bf16(fw.z, Bh[r][c + 2], Bl[r][c + 2]);
      split_bf16(fw.w, Bh[r][c + 3], Bl[r][c + 3]);
    }
    __syncthreads();
    mma_block(Ah, Al, Bh, Bl, wm, wn, lane, acc);
  }

  const int nloc = lane & 15, mhalf = (lane >> 4) << 3;
#pragma unroll
  for (int mt = 0; mt < 4; ++mt)
#pragma unroll
    for (int nt = 0; nt < 2; ++nt)
#pragma unroll
      for (int v = 0; v < 8; ++v)
        C[(m0 + 64 * wm + 16 * mt + mhalf + v) * (size_t)N + n0 + 32 * wn + 16 * nt + nloc] =
            acc[mt][nt][v];
}

// ---------------------------------------------------------------------------
// Kernel 2: depthwise 3-tap conv along frames; emits bf16 hi/lo planes.
// mode 0: (bc,band,w,d) [Q,K];  mode 1: (bc,band,d,w) [V transposed]
// ---------------------------------------------------------------------------
__global__ __launch_bounds__(256) void dwconv_kernel(const float *__restrict__ Y,
                                                     const float *__restrict__ cw,
                                                     const float *__restrict__ cb,
                                                     unsigned short *__restrict__ Oh,
                                                     unsigned short *__restrict__ Ol, int mode) {
  size_t idx = (size_t)blockIdx.x * 256 + threadIdx.x;  // 8*512*2048 = 8388608 total
  int h = (int)(idx & 2047);
  size_t rest = idx >> 11;
  int wpos = (int)(rest & 511);
  size_t bc = rest >> 9;
  float w0 = cw[h * 3 + 0], w1 = cw[h * 3 + 1], w2 = cw[h * 3 + 2];
  float ym = (wpos > 0) ? Y[idx - 2048] : 0.0f;
  float y0 = Y[idx];
  float yp = (wpos < 511) ? Y[idx + 2048] : 0.0f;
  float r = fmaf(ym, w0, fmaf(y0, w1, fmaf(yp, w2, cb[h])));
  int band = h >> 7, dd = h & 127;
  size_t o = (mode == 0) ? (((bc * 16 + band) * 512 + wpos) * 128 + dd)
                         : (((bc * 16 + band) * 128 + dd) * 512 + wpos);
  unsigned short hi, lo;
  split_bf16(r, hi, lo);
  Oh[o] = hi;
  Ol[o] = lo;
}

// rel_t[i][d] = rel_w[d][i], bf16 hi/lo
__global__ __launch_bounds__(256) void relT_kernel(const float *__restrict__ rel,
                                                   unsigned short *__restrict__ Rh,
                                                   unsigned short *__restrict__ Rl) {
  int idx = blockIdx.x * 256 + threadIdx.x;  // 512*128
  int dd = idx & 127, i = idx >> 7;
  unsigned short hi, lo;
  split_bf16(rel[dd * 512 + i], hi, lo);
  Rh[i * 128 + dd] = hi;
  Rl[i * 128 + dd] = lo;
}

// ---------------------------------------------------------------------------
// Kernel 3: logits  S = [Q | relT] . [K | Q]^T  (K=256), epilogue scale + prev_qk
// grid: (4, 4, 128 batches = bc*band)
// ---------------------------------------------------------------------------
__global__ __launch_bounds__(256) void attn_logits_kernel(
    const unsigned short *__restrict__ Qh, const unsigned short *__restrict__ Ql,
    const unsigned short *__restrict__ Kh, const unsigned short *__restrict__ Kl,
    const unsigned short *__restrict__ Rh, const unsigned short *__restrict__ Rl,
    const float *__restrict__ prev, float *__restrict__ qkOut) {
  __shared__ __align__(8) unsigned short Ah[128][34], Al[128][34], Bh[128][34], Bl[128][34];
  const int tid = threadIdx.x, lane = tid & 31, wid = tid >> 5;
  const int wm = wid & 1, wn = wid >> 1;
  const int batch = blockIdx.z;
  const size_t m0 = (size_t)blockIdx.y * 128, n0 = (size_t)blockIdx.x * 128;
  v8f acc[4][2];
  zero_acc(acc);

  for (int kt = 0; kt < 256; kt += 32) {
    __syncthreads();
#pragma unroll 1
    for (int p = tid; p < 128 * 16; p += 256) {
      int r = p >> 4, c = (p & 15) << 1;
      const unsigned short *pah, *pal, *pbh, *pbl;
      if (kt < 128) {
        size_t ia = ((size_t)batch * 512 + m0 + r) * 128 + kt + c;
        size_t ib = ((size_t)batch * 512 + n0 + r) * 128 + kt + c;
        pah = Qh + ia; pal = Ql + ia; pbh = Kh + ib; pbl = Kl + ib;
      } else {
        size_t ia = (m0 + r) * 128 + (size_t)(kt - 128) + c;
        size_t ib = ((size_t)batch * 512 + n0 + r) * 128 + (kt - 128) + c;
        pah = Rh + ia; pal = Rl + ia; pbh = Qh + ib; pbl = Ql + ib;
      }
      *(unsigned *)&Ah[r][c] = *(const unsigned *)pah;
      *(unsigned *)&Al[r][c] = *(const unsigned *)pal;
      *(unsigned *)&Bh[r][c] = *(const unsigned *)pbh;
      *(unsigned *)&Bl[r][c] = *(const unsigned *)pbl;
    }
    __syncthreads();
    mma_block(Ah, Al, Bh, Bl, wm, wn, lane, acc);
  }

  const float scale = 0.022097086912079608f;  // 1/sqrt(2048)
  const int nloc = lane & 15, mhalf = (lane >> 4) << 3;
#pragma unroll
  for (int mt = 0; mt < 4; ++mt)
#pragma unroll
    for (int nt = 0; nt < 2; ++nt)
#pragma unroll
      for (int v = 0; v < 8; ++v) {
        size_t row = m0 + 64 * wm + 16 * mt + mhalf + v;
        size_t col = n0 + 32 * wn + 16 * nt + nloc;
        size_t o = ((size_t)batch * 512 + row) * 512 + col;
        qkOut[o] = fmaf(acc[mt][nt][v], scale, prev[o]);
      }
}

// ---------------------------------------------------------------------------
// Kernel 4: row softmax over 512, emits P as bf16 hi/lo. One wave per row.
// ---------------------------------------------------------------------------
__global__ __launch_bounds__(256) void softmax_kernel(const float *__restrict__ qk,
                                                      unsigned short *__restrict__ Ph,
                                                      unsigned short *__restrict__ Pl) {
  const int row = blockIdx.x * 8 + (threadIdx.x >> 5);
  const int lane = threadIdx.x & 31;
  const float *src = qk + (size_t)row * 512;
  float vals[16];
  float mx = -3.4e38f;
#pragma unroll
  for (int t = 0; t < 16; ++t) {
    vals[t] = src[lane + t * 32];
    mx = fmaxf(mx, vals[t]);
  }
#pragma unroll
  for (int m = 16; m >= 1; m >>= 1) mx = fmaxf(mx, __shfl_xor(mx, m, 32));
  float sum = 0.0f;
#pragma unroll
  for (int t = 0; t < 16; ++t) {
    vals[t] = __expf(vals[t] - mx);
    sum += vals[t];
  }
#pragma unroll
  for (int m = 16; m >= 1; m >>= 1) sum += __shfl_xor(sum, m, 32);
  const float inv = 1.0f / sum;
#pragma unroll
  for (int t = 0; t < 16; ++t) {
    unsigned short hi, lo;
    split_bf16(vals[t] * inv, hi, lo);
    size_t o = (size_t)row * 512 + lane + t * 32;
    Ph[o] = hi;
    Pl[o] = lo;
  }
}

// ---------------------------------------------------------------------------
// Kernel 5: A = P.V  (per batch 512x128, K=512); writes Aout in (bc,w,h) layout.
// grid: (1, 4, 128)
// ---------------------------------------------------------------------------
__global__ __launch_bounds__(256) void pv_kernel(const unsigned short *__restrict__ Ph,
                                                 const unsigned short *__restrict__ Pl,
                                                 const unsigned short *__restrict__ Vh,
                                                 const unsigned short *__restrict__ Vl,
                                                 float *__restrict__ Aout) {
  __shared__ __align__(8) unsigned short Ah[128][34], Al[128][34], Bh[128][34], Bl[128][34];
  const int tid = threadIdx.x, lane = tid & 31, wid = tid >> 5;
  const int wm = wid & 1, wn = wid >> 1;
  const int batch = blockIdx.z;
  const size_t m0 = (size_t)blockIdx.y * 128;
  v8f acc[4][2];
  zero_acc(acc);

  for (int kt = 0; kt < 512; kt += 32) {
    __syncthreads();
#pragma unroll 1
    for (int p = tid; p < 128 * 16; p += 256) {
      int r = p >> 4, c = (p & 15) << 1;
      size_t ia = ((size_t)batch * 512 + m0 + r) * 512 + kt + c;  // P: (batch,i,j)
      size_t ib = ((size_t)batch * 128 + r) * 512 + kt + c;       // V^T: (batch,d,j)
      *(unsigned *)&Ah[r][c] = *(const unsigned *)(Ph + ia);
      *(unsigned *)&Al[r][c] = *(const unsigned *)(Pl + ia);
      *(unsigned *)&Bh[r][c] = *(const unsigned *)(Vh + ib);
      *(unsigned *)&Bl[r][c] = *(const unsigned *)(Vl + ib);
    }
    __syncthreads();
    mma_block(Ah, Al, Bh, Bl, wm, wn, lane, acc);
  }

  const int bc = batch >> 4, band = batch & 15;
  const int nloc = lane & 15, mhalf = (lane >> 4) << 3;
#pragma unroll
  for (int mt = 0; mt < 4; ++mt)
#pragma unroll
    for (int nt = 0; nt < 2; ++nt)
#pragma unroll
      for (int v = 0; v < 8; ++v) {
        size_t i = m0 + 64 * wm + 16 * mt + mhalf + v;
        int dd = 32 * wn + 16 * nt + nloc;
        Aout[((size_t)bc * 512 + i) * 2048 + band * 128 + dd] = acc[mt][nt][v];
      }
}

// ---------------------------------------------------------------------------
extern "C" void kernel_launch(void *const *d_in, const int *in_sizes, int n_in, void *d_out,
                              int out_size, void *d_ws, size_t ws_size, hipStream_t stream) {
  (void)in_sizes; (void)n_in; (void)out_size; (void)ws_size;
  const float *x    = (const float *)d_in[0];
  const float *prev = (const float *)d_in[1];
  const float *Wq   = (const float *)d_in[2];
  const float *Wk   = (const float *)d_in[3];
  const float *Wv   = (const float *)d_in[4];
  const float *Wo   = (const float *)d_in[5];
  const float *cqw  = (const float *)d_in[6];
  const float *cqb  = (const float *)d_in[7];
  const float *ckw  = (const float *)d_in[8];
  const float *ckb  = (const float *)d_in[9];
  const float *cvw  = (const float *)d_in[10];
  const float *cvb  = (const float *)d_in[11];
  const float *rel  = (const float *)d_in[12];

  float *oOut  = (float *)d_out;               // 8,388,608 floats
  float *qkOut = (float *)d_out + 8388608ull;  // 33,554,432 floats

  char *ws = (char *)d_ws;
  float *Yq = (float *)(ws + 0ull);
  float *Yk = (float *)(ws + 33554432ull);
  float *Yv = (float *)(ws + 67108864ull);
  unsigned short *Qh = (unsigned short *)(ws + 100663296ull);
  unsigned short *Ql = (unsigned short *)(ws + 117440512ull);
  unsigned short *Kh = (unsigned short *)(ws + 134217728ull);
  unsigned short *Kl = (unsigned short *)(ws + 150994944ull);
  unsigned short *Vh = (unsigned short *)(ws + 167772160ull);
  unsigned short *Vl = (unsigned short *)(ws + 184549376ull);
  unsigned short *Rh = (unsigned short *)(ws + 201326592ull);
  unsigned short *Rl = (unsigned short *)(ws + 201457664ull);
  unsigned short *Ph = (unsigned short *)(ws + 201588736ull);
  unsigned short *Pl = (unsigned short *)(ws + 268697600ull);
  float *Aout = Yq;  // Y buffers are dead after dwconv; reuse for A

  // rel^T bf16 prep
  relT_kernel<<<256, 256, 0, stream>>>(rel, Rh, Rl);

  // Q0/K0/V0 = X . W{q,k,v}^T  (M=4096, N=2048, K=2048), one launch, z picks matrix
  gemm_xwT_kernel<<<dim3(16, 32, 3), 256, 0, stream>>>(x, Wq, Wk, Wv, Yq, Yk, Yv, 2048, 2048);

  // depthwise conv along frames -> bf16 hi/lo planes in attention layouts
  dwconv_kernel<<<32768, 256, 0, stream>>>(Yq, cqw, cqb, Qh, Ql, 0);
  dwconv_kernel<<<32768, 256, 0, stream>>>(Yk, ckw, ckb, Kh, Kl, 0);
  dwconv_kernel<<<32768, 256, 0, stream>>>(Yv, cvw, cvb, Vh, Vl, 1);

  // qk = ([Q|relT].[K|Q]^T)/sqrt(h) + prev_qk   (second output)
  attn_logits_kernel<<<dim3(4, 4, 128), 256, 0, stream>>>(Qh, Ql, Kh, Kl, Rh, Rl, prev, qkOut);

  // row softmax -> P (bf16 hi/lo)
  softmax_kernel<<<8192, 256, 0, stream>>>(qkOut, Ph, Pl);

  // A = P.V, written directly in (bc, w, h) layout
  pv_kernel<<<dim3(1, 4, 128), 256, 0, stream>>>(Ph, Pl, Vh, Vl, Aout);

  // o = A . Wo^T  (first output)
  gemm_xwT_kernel<<<dim3(16, 32, 1), 256, 0, stream>>>(Aout, Wo, Wo, Wo, oOut, oOut, oOut, 2048,
                                                       2048);
}
